// LocalInteraction_10995116278396
// MI455X (gfx1250) — matrix operand order repaired and checked by reading, not compiled, not run on hardware
//
#include <hip/hip_runtime.h>

// ---------------------------------------------------------------------------
// LocalInteraction for MI455X (gfx1250, wave32, WMMA f32_16x16x32_f16)
//
// d_in flatten assumption (setup_inputs dict order, params in insertion order):
//  0:x 1:rbf 2:pij 3:dij 4:idx_i 5:idx_j 6:Ws 7:Wp 8:Wd 9:Pp 10:Pd
//  11..16: mlp_x {W1,b1,W2,b2,W,b}  17..22: mlp_s  23..28: mlp_p
//  29..34: mlp_d  35..40: mlp_out
// ---------------------------------------------------------------------------

static constexpr int FF = 64;   // feature dim
static constexpr int RR = 32;   // rbf dim
static constexpr int NN = 50000;
static constexpr int PP = 400000;

typedef __attribute__((ext_vector_type(16))) _Float16 v16h;
typedef __attribute__((ext_vector_type(8)))  _Float16 v8h;
typedef __attribute__((ext_vector_type(8)))  float    v8f;

struct MlpW { const float *W1, *b1, *W2, *b2, *W, *b; };

__device__ __forceinline__ v8f wmma32(v16h a, v16h b, v8f c) {
  // D(16x16,f32) = A(16x32,f16) x B(32x16,f16) + C
  return __builtin_amdgcn_wmma_f32_16x16x32_f16(false, a, false, b, (short)0, c,
                                                false, false);
}

__device__ __forceinline__ float silu_f(float x) {
  // x * sigmoid(x); v_exp_f32 + v_rcp_f32 (avoid IEEE div expansion)
  return x * __builtin_amdgcn_rcpf(1.0f + __expf(-x));
}

// A fragment (16x32) from row-major f32 source; rowp = this lane's row.
// a[i]   = row[k0 + hi*8 + i]       (K = k0 .. k0+15 region)
// a[8+i] = row[k0 + 16 + hi*8 + i]  (K = k0+16 .. k0+31 region)
__device__ __forceinline__ v16h a_frag_f32(const float* rowp, int k0, int hi) {
  v16h a;
  const float* p0 = rowp + k0 + hi * 8;
  const float* p1 = p0 + 16;
#pragma unroll
  for (int i = 0; i < 8; ++i) { a[i] = (_Float16)p0[i]; a[8 + i] = (_Float16)p1[i]; }
  return a;
}

__device__ __forceinline__ v16h a_frag_h(const _Float16* rowp, int k0, int hi) {
  v16h a;
  const _Float16* p0 = rowp + k0 + hi * 8;
  const _Float16* p1 = p0 + 16;
#pragma unroll
  for (int i = 0; i < 8; ++i) { a[i] = p0[i]; a[8 + i] = p1[i]; }
  return a;
}

// B fragment (32x16) from f16 row-major weight W[n][k] (ld = row length).
// B[k,n] = W[n,k]  ==>  computes  X @ W^T.  b[i]=k0+hi*8+i, b[8+i]=k0+16+hi*8+i.
__device__ __forceinline__ v16h b_frag_h(const _Float16* W, int ld, int nt,
                                         int k0, int lane) {
  int n  = nt * 16 + (lane & 15);
  int hi = (lane >> 4) & 1;
  v16h b;
  const _Float16* p = W + n * ld + k0 + hi * 8;
#pragma unroll
  for (int i = 0; i < 8; ++i) { b[i] = p[i]; b[8 + i] = p[16 + i]; }
  return b;
}

// Write 4 D-layout tiles (16 rows x 64 cols) through silu() into f16 LDS
// staging (row-major 16x64) so they can be re-read as A fragments.
__device__ __forceinline__ void stage_tiles_silu(_Float16* st, const v8f* t,
                                                 int lane) {
  int m = lane & 15, hi = (lane >> 4) & 1;
#pragma unroll
  for (int nt = 0; nt < 4; ++nt)
#pragma unroll
    for (int r = 0; r < 8; ++r)
      st[(r + hi * 8) * FF + nt * 16 + m] = (_Float16)silu_f(t[nt][r]);
}

// One SpookyNet residual MLP (NRES=1) on a 16-node tile.
//   out = silu(x + silu(silu(x)@W1^T + b1)@W2^T + b2) @ W^T + b
// a0x/a1x are precomputed A fragments of silu(x) (shared across MLPs).
// Biases / residual are folded into the WMMA C initializer (no post-WMMA
// dependent VALU -> no hazard NOPs).
__device__ __forceinline__ void run_mlp(v16h a0x, v16h a1x, const v8f xin[4],
                                        const _Float16* W1h,
                                        const _Float16* W2h, const _Float16* Woh,
                                        const float* b1, const float* b2,
                                        const float* bo, _Float16* st,
                                        float* __restrict__ out, int row0,
                                        int N, int lane) {
  int m = lane & 15, hi = (lane >> 4) & 1;
  v8f h1[4];
#pragma unroll
  for (int nt = 0; nt < 4; ++nt) {
    float bb = b1[nt * 16 + m];
    v8f c;
#pragma unroll
    for (int r = 0; r < 8; ++r) c[r] = bb;   // bias rides the matrix pipe
    c = wmma32(a0x, b_frag_h(W1h, FF, nt, 0, lane), c);
    c = wmma32(a1x, b_frag_h(W1h, FF, nt, 32, lane), c);
    h1[nt] = c;
  }
  stage_tiles_silu(st, h1, lane);
  __builtin_amdgcn_wave_barrier();
  v16h a0 = a_frag_h(st + m * FF, 0, hi);
  v16h a1 = a_frag_h(st + m * FF, 32, hi);
  v8f u[4];
#pragma unroll
  for (int nt = 0; nt < 4; ++nt) {
    float bb = b2[nt * 16 + m];
    v8f c;
#pragma unroll
    for (int r = 0; r < 8; ++r) c[r] = xin[nt][r] + bb;  // residual + bias in C
    c = wmma32(a0, b_frag_h(W2h, FF, nt, 0, lane), c);
    c = wmma32(a1, b_frag_h(W2h, FF, nt, 32, lane), c);
    u[nt] = c;
  }
  stage_tiles_silu(st, u, lane);
  __builtin_amdgcn_wave_barrier();
  a0 = a_frag_h(st + m * FF, 0, hi);
  a1 = a_frag_h(st + m * FF, 32, hi);
#pragma unroll
  for (int nt = 0; nt < 4; ++nt) {
    float bb = bo[nt * 16 + m];
    v8f c;
#pragma unroll
    for (int r = 0; r < 8; ++r) c[r] = bb;
    c = wmma32(a0, b_frag_h(Woh, FF, nt, 0, lane), c);
    c = wmma32(a1, b_frag_h(Woh, FF, nt, 32, lane), c);
#pragma unroll
    for (int r = 0; r < 8; ++r) {
      int rr = row0 + r + hi * 8;
      if (rr < N) out[rr * FF + nt * 16 + m] = c[r];
    }
  }
}

// ---------------------------------------------------------------------------
// Kernel 1: four input-side residual MLPs share silu(x); one wave = 16 nodes.
// Weights converted f32->f16 into LDS once per block per MLP phase.
// ---------------------------------------------------------------------------
__global__ void __launch_bounds__(256)
node_mlps_kernel(const float* __restrict__ x, MlpW m0, MlpW m1, MlpW m2,
                 MlpW m3, float* o0, float* o1, float* o2, float* o3, int N) {
  __shared__ __align__(16) _Float16 wbuf[3 * FF * FF];   // 24 KB
  __shared__ __align__(16) _Float16 stage[8 * 16 * FF];  // 16 KB
  int wave = threadIdx.x >> 5, lane = threadIdx.x & 31;
  int m = lane & 15, hi = (lane >> 4) & 1;
  int row0 = (blockIdx.x * 8 + wave) * 16;
  bool act = row0 < N;
  _Float16* st = &stage[wave * 16 * FF];

  v8f xt[4] = {};
  v16h a0x = {}, a1x = {};
  if (act) {
#pragma unroll
    for (int nt = 0; nt < 4; ++nt)
#pragma unroll
      for (int r = 0; r < 8; ++r) {
        int rr = min(row0 + r + hi * 8, N - 1);
        xt[nt][r] = x[rr * FF + nt * 16 + m];
      }
    // silu(x) A fragments built once, reused by all four MLPs
    stage_tiles_silu(st, xt, lane);
    __builtin_amdgcn_wave_barrier();
    a0x = a_frag_h(st + m * FF, 0, hi);
    a1x = a_frag_h(st + m * FF, 32, hi);
  }
  const MlpW mlps[4] = {m0, m1, m2, m3};
  float* const outs[4] = {o0, o1, o2, o3};
#pragma unroll
  for (int p = 0; p < 4; ++p) {
    __syncthreads();
    const float* srcs[3] = {mlps[p].W1, mlps[p].W2, mlps[p].W};
#pragma unroll
    for (int w = 0; w < 3; ++w)
      for (int i = threadIdx.x; i < FF * FF; i += blockDim.x)
        wbuf[w * FF * FF + i] = (_Float16)srcs[w][i];
    __syncthreads();
    if (act)
      run_mlp(a0x, a1x, xt, wbuf, wbuf + FF * FF, wbuf + 2 * FF * FF,
              mlps[p].b1, mlps[p].b2, mlps[p].b, st, outs[p], row0, N, lane);
  }
}

// ---------------------------------------------------------------------------
// Kernel 1b: CSR row starts via binary search on sorted idx_i.
// ---------------------------------------------------------------------------
__global__ void __launch_bounds__(256)
row_start_kernel(const int* __restrict__ idx_i, int* __restrict__ row_start,
                 int N, int P) {
  int n = blockIdx.x * blockDim.x + threadIdx.x;
  if (n > N) return;
  int lo = 0, hib = P;
  while (lo < hib) {
    int mid = (lo + hib) >> 1;
    if (idx_i[mid] < n) lo = mid + 1; else hib = mid;
  }
  row_start[n] = lo;
}

// ---------------------------------------------------------------------------
// Kernel 2: fused radial GEMM + per-pair products + atomic-free segment sum.
// One wave owns 16 destination nodes (a contiguous, sorted pair range) and
// walks it 32 pairs at a time.  The scatter is a WMMA with a 0/1 membership
// A-matrix:  acc[16n x 16f] += M[16n x 32pair] x V[32pair x 16f].
// MODE 0: channels {s, p0..p2} (Ws/Xs, Wp/Xp*pij);  MODE 1: {d0..d4}.
// ---------------------------------------------------------------------------
template <int MODE>
__global__ void __launch_bounds__(64)
pair_aggregate_kernel(const float* __restrict__ rbf, const float* __restrict__ vec,
                      const int* __restrict__ idx_i, const int* __restrict__ idx_j,
                      const float* __restrict__ Wa, const float* __restrict__ Wb,
                      const float* __restrict__ Xa, const float* __restrict__ Xb,
                      const int* __restrict__ row_start,
                      float* __restrict__ OutA, float* __restrict__ OutB,
                      int N, int P) {
  constexpr int NCH  = (MODE == 0) ? 4 : 5;
  constexpr int NVEC = (MODE == 0) ? 3 : 5;
  constexpr int NWV  = 2;
  __shared__ __align__(16) _Float16 wlds[(MODE == 0 ? 2 : 1) * FF * RR];
  __shared__ __align__(16) _Float16 stage[NWV * NCH * FF * RR];  // col-major [n][k]
  __shared__ int   ii_s[NWV * 32];
  __shared__ int   jj_s[NWV * 32];
  __shared__ float vv_s[NWV * 32 * NVEC];

  for (int i = threadIdx.x; i < FF * RR; i += blockDim.x)
    wlds[i] = (_Float16)Wa[i];
  if (MODE == 0)
    for (int i = threadIdx.x; i < FF * RR; i += blockDim.x)
      wlds[FF * RR + i] = (_Float16)Wb[i];
  __syncthreads();

  int wave = threadIdx.x >> 5, lane = threadIdx.x & 31;
  int m = lane & 15, hi = (lane >> 4) & 1;
  int n0 = (blockIdx.x * NWV + wave) * 16;
  if (n0 >= N) return;
  int node = n0 + m;
  int pbeg = row_start[n0];
  int pend = row_start[min(n0 + 16, N)];

  _Float16* stg = &stage[wave * NCH * FF * RR];
  int*   ii = &ii_s[wave * 32];
  int*   jj = &jj_s[wave * 32];
  float* vv = &vv_s[wave * 32 * NVEC];

  v8f acc[NCH][4];
#pragma unroll
  for (int c = 0; c < NCH; ++c)
#pragma unroll
    for (int nt = 0; nt < 4; ++nt) acc[c][nt] = {};

  for (int e0 = pbeg; e0 < pend; e0 += 32) {
    // prefetch next chunk's rbf rows into the cache hierarchy
    if (e0 + 32 < pend)
      __builtin_prefetch(rbf + min(e0 + 32 + lane, P - 1) * RR, 0, 1);

    // stage 32 pairs of metadata (wave-synchronous, wave-private LDS)
    {
      int e = e0 + lane;
      bool v = e < pend;
      int ec = min(e, P - 1);
      ii[lane] = v ? idx_i[ec] : -1;  // -1 => membership 0 for tail pairs
      jj[lane] = v ? (idx_j[ec] * FF) : 0;  // pre-scaled gather offset
#pragma unroll
      for (int c = 0; c < NVEC; ++c)
        vv[lane * NVEC + c] = v ? vec[ec * NVEC + c] : 0.0f;
    }
    __builtin_amdgcn_wave_barrier();

    // membership A fragment: A[m][k] = (idx_i[e0+k] == node)
    v16h ma;
#pragma unroll
    for (int i = 0; i < 8; ++i) {
      ma[i]     = (_Float16)((ii[hi * 8 + i]      == node) ? 1.0f : 0.0f);
      ma[8 + i] = (_Float16)((ii[16 + hi * 8 + i] == node) ? 1.0f : 0.0f);
    }
    // rbf A fragments for the two 16-pair sub-tiles (K = 32 rbf features)
    v16h ar0 = a_frag_f32(rbf + min(e0 + m,      P - 1) * RR, 0, hi);
    v16h ar1 = a_frag_f32(rbf + min(e0 + 16 + m, P - 1) * RR, 0, hi);

#pragma unroll
    for (int t = 0; t < 2; ++t) {
      v16h ar = t ? ar1 : ar0;
#pragma unroll
      for (int nt = 0; nt < 4; ++nt) {
        v8f z = {};
        v8f g0 = wmma32(ar, b_frag_h(wlds, RR, nt, 0, lane), z);  // rbf@Wa^T
        v8f g1 = {};
        if (MODE == 0)
          g1 = wmma32(ar, b_frag_h(wlds + FF * RR, RR, nt, 0, lane), z);
        int col = nt * 16 + m;
        v8h o[NCH];
#pragma unroll
        for (int r = 0; r < 8; ++r) {
          int kr = t * 16 + hi * 8 + r;   // pair row within chunk
          int jo = jj[kr];                // idx_j * FF
          if (MODE == 0) {
            float xs = Xa[jo + col];
            float xp = Xb[jo + col];
            o[0][r] = (_Float16)(g0[r] * xs);
            float gp = g1[r] * xp;
            o[1][r] = (_Float16)(gp * vv[kr * NVEC + 0]);
            o[2][r] = (_Float16)(gp * vv[kr * NVEC + 1]);
            o[3][r] = (_Float16)(gp * vv[kr * NVEC + 2]);
          } else {
            float gd = g0[r] * Xa[jo + col];
#pragma unroll
            for (int c = 0; c < NCH; ++c)
              o[c][r] = (_Float16)(gd * vv[kr * NVEC + c]);
          }
        }
        int soff = col * RR + t * 16 + hi * 8;  // col-major, 16B-aligned
#pragma unroll
        for (int c = 0; c < NCH; ++c)
          *(v8h*)(stg + c * FF * RR + soff) = o[c];
      }
    }
    __builtin_amdgcn_wave_barrier();
    // segment-sum = membership x values
#pragma unroll
    for (int c = 0; c < NCH; ++c)
#pragma unroll
      for (int nt = 0; nt < 4; ++nt)
        acc[c][nt] = wmma32(ma, b_frag_h(stg + c * FF * RR, RR, nt, 0, lane),
                            acc[c][nt]);
  }

  // exclusive ownership of these 16 nodes -> plain stores, no atomics
#pragma unroll
  for (int nt = 0; nt < 4; ++nt) {
    int col = nt * 16 + m;
#pragma unroll
    for (int r = 0; r < 8; ++r) {
      int n = n0 + r + hi * 8;
      if (n < N) {
        if (MODE == 0) {
          OutA[n * FF + col] = acc[0][nt][r];
#pragma unroll
          for (int c = 0; c < 3; ++c)
            OutB[(n * 3 + c) * FF + col] = acc[c + 1][nt][r];
        } else {
#pragma unroll
          for (int c = 0; c < 5; ++c)
            OutA[(n * 5 + c) * FF + col] = acc[c][nt][r];
        }
      }
    }
  }
}

// ---------------------------------------------------------------------------
// Kernel 3: pa/pb + da/db projections, combine, output residual MLP.
// ---------------------------------------------------------------------------
__global__ void __launch_bounds__(256)
project_out_kernel(const float* __restrict__ Sseg, const float* __restrict__ Pseg,
                   const float* __restrict__ Dseg, const float* __restrict__ XX,
                   const float* __restrict__ Pp, const float* __restrict__ Pd,
                   MlpW mo, float* __restrict__ out, int N) {
  __shared__ __align__(16) _Float16 wbuf[3 * FF * FF];   // 24 KB, phase-reused
  __shared__ __align__(16) _Float16 stage[8 * 16 * FF];  // 16 KB
  int wave = threadIdx.x >> 5, lane = threadIdx.x & 31;
  int m = lane & 15, hi = (lane >> 4) & 1;
  int n0 = (blockIdx.x * 8 + wave) * 16;
  bool act = n0 < N;
  _Float16* st = &stage[wave * 16 * FF];

  v8f accq[4] = {};
  int row = min(n0 + m, N - 1);

  // phase P: (p @ Pp^T) -> pa*pb summed over the 3 axis
  for (int i = threadIdx.x; i < 2 * FF * FF; i += blockDim.x)
    wbuf[i] = (_Float16)Pp[i];
  __syncthreads();
  if (act) {
#pragma unroll
    for (int c = 0; c < 3; ++c) {
      const float* rp = Pseg + (row * 3 + c) * FF;
      v16h a0 = a_frag_f32(rp, 0, hi);
      v16h a1 = a_frag_f32(rp, 32, hi);
      v8f y[8];
#pragma unroll
      for (int nt = 0; nt < 8; ++nt) {   // 128 output cols = pa | pb
        v8f cc = {};
        cc = wmma32(a0, b_frag_h(wbuf, FF, nt, 0, lane), cc);
        cc = wmma32(a1, b_frag_h(wbuf, FF, nt, 32, lane), cc);
        y[nt] = cc;
      }
#pragma unroll
      for (int q = 0; q < 4; ++q)
#pragma unroll
        for (int r = 0; r < 8; ++r) accq[q][r] += y[q][r] * y[q + 4][r];
    }
  }
  __syncthreads();
  // phase D
  for (int i = threadIdx.x; i < 2 * FF * FF; i += blockDim.x)
    wbuf[i] = (_Float16)Pd[i];
  __syncthreads();
  if (act) {
#pragma unroll
    for (int c = 0; c < 5; ++c) {
      const float* rp = Dseg + (row * 5 + c) * FF;
      v16h a0 = a_frag_f32(rp, 0, hi);
      v16h a1 = a_frag_f32(rp, 32, hi);
      v8f y[8];
#pragma unroll
      for (int nt = 0; nt < 8; ++nt) {
        v8f cc = {};
        cc = wmma32(a0, b_frag_h(wbuf, FF, nt, 0, lane), cc);
        cc = wmma32(a1, b_frag_h(wbuf, FF, nt, 32, lane), cc);
        y[nt] = cc;
      }
#pragma unroll
      for (int q = 0; q < 4; ++q)
#pragma unroll
        for (int r = 0; r < 8; ++r) accq[q][r] += y[q][r] * y[q + 4][r];
    }
  }
  __syncthreads();
  // phase out-MLP
  {
    const float* srcs[3] = {mo.W1, mo.W2, mo.W};
#pragma unroll
    for (int w = 0; w < 3; ++w)
      for (int i = threadIdx.x; i < FF * FF; i += blockDim.x)
        wbuf[w * FF * FF + i] = (_Float16)srcs[w][i];
  }
  __syncthreads();
  if (act) {
    v8f u[4];
#pragma unroll
    for (int nt = 0; nt < 4; ++nt) {
      int col = nt * 16 + m;
#pragma unroll
      for (int r = 0; r < 8; ++r) {
        int n = min(n0 + r + hi * 8, N - 1);
        u[nt][r] = Sseg[n * FF + col] + XX[n * FF + col] + accq[nt][r];
      }
    }
    stage_tiles_silu(st, u, lane);
    __builtin_amdgcn_wave_barrier();
    v16h a0u = a_frag_h(st + m * FF, 0, hi);
    v16h a1u = a_frag_h(st + m * FF, 32, hi);
    run_mlp(a0u, a1u, u, wbuf, wbuf + FF * FF, wbuf + 2 * FF * FF, mo.b1,
            mo.b2, mo.b, st, out, n0, N, lane);
  }
}

// ---------------------------------------------------------------------------
extern "C" void kernel_launch(void* const* d_in, const int* in_sizes, int n_in,
                              void* d_out, int out_size, void* d_ws,
                              size_t ws_size, hipStream_t stream) {
  (void)in_sizes; (void)n_in; (void)out_size; (void)ws_size;
  const int N = NN, P = PP;
  const float* x     = (const float*)d_in[0];
  const float* rbf   = (const float*)d_in[1];
  const float* pij   = (const float*)d_in[2];
  const float* dij   = (const float*)d_in[3];
  const int*   idx_i = (const int*)d_in[4];
  const int*   idx_j = (const int*)d_in[5];
  const float* Ws    = (const float*)d_in[6];
  const float* Wp    = (const float*)d_in[7];
  const float* Wd    = (const float*)d_in[8];
  const float* Pp    = (const float*)d_in[9];
  const float* Pd    = (const float*)d_in[10];
  auto mlp = [&](int b) {
    MlpW w;
    w.W1 = (const float*)d_in[b];     w.b1 = (const float*)d_in[b + 1];
    w.W2 = (const float*)d_in[b + 2]; w.b2 = (const float*)d_in[b + 3];
    w.W  = (const float*)d_in[b + 4]; w.b  = (const float*)d_in[b + 5];
    return w;
  };
  MlpW mx = mlp(11), ms = mlp(17), mp = mlp(23), md = mlp(29), mo = mlp(35);

  // workspace: 13*N*F floats + (N+1) ints  (~167 MB)
  float* wsf  = (float*)d_ws;
  float* XX   = wsf;
  float* XS   = XX + (size_t)N * FF;
  float* XP   = XS + (size_t)N * FF;
  float* XD   = XP + (size_t)N * FF;
  float* Sseg = XD + (size_t)N * FF;
  float* Pseg = Sseg + (size_t)N * FF;
  float* Dseg = Pseg + (size_t)N * 3 * FF;
  int*   rs   = (int*)(Dseg + (size_t)N * 5 * FF);

  const int tiles = (N + 15) / 16;  // 3125
  node_mlps_kernel<<<(tiles + 7) / 8, 256, 0, stream>>>(x, mx, ms, mp, md, XX,
                                                        XS, XP, XD, N);
  row_start_kernel<<<(N + 1 + 255) / 256, 256, 0, stream>>>(idx_i, rs, N, P);
  pair_aggregate_kernel<0><<<(tiles + 1) / 2, 64, 0, stream>>>(
      rbf, pij, idx_i, idx_j, Ws, Wp, XS, XP, rs, Sseg, Pseg, N, P);
  pair_aggregate_kernel<1><<<(tiles + 1) / 2, 64, 0, stream>>>(
      rbf, dij, idx_i, idx_j, Wd, nullptr, XD, nullptr, rs, Dseg, nullptr, N, P);
  project_out_kernel<<<(tiles + 7) / 8, 256, 0, stream>>>(
      Sseg, Pseg, Dseg, XX, Pp, Pd, mo, (float*)d_out, N);
}
